// GroupedQueryAttention_59966333386980
// MI455X (gfx1250) — matrix operand order repaired and checked by reading, not compiled, not run on hardware
//
#include <hip/hip_runtime.h>

// ---------------------------------------------------------------------------
// GQA attention block for MI455X (gfx1250, wave32, WMMA 16x16x32 bf16).
// Pipeline: fp32->bf16 pack, QKV GEMMs (WMMA), RoPE/transpose, flash-attn
// (WMMA scores + WMMA p@v with online softmax), output-projection GEMM.
// Softmax row reductions use DPP16 butterflies (no LDS/bpermute traffic).
// ---------------------------------------------------------------------------

typedef unsigned short u16;
typedef __attribute__((ext_vector_type(16))) __bf16 v16bf;
typedef __attribute__((ext_vector_type(8)))  __bf16 v8bf;
typedef __attribute__((ext_vector_type(8)))  float  v8f;

#define DEV __device__ __forceinline__

// fp32 -> bf16 round-to-nearest-even
DEV u16 f2b(float f) {
  union { float f; unsigned u; } x; x.f = f;
  unsigned u = x.u;
  return (u16)((u + 0x7FFFu + ((u >> 16) & 1u)) >> 16);
}
DEV float b2f(u16 h) {
  union { unsigned u; float f; } x; x.u = ((unsigned)h) << 16; return x.f;
}
DEV int   f2i(float f) { union { float f; int i; } u; u.f = f; return u.i; }
DEV float i2f(int i)   { union { int i; float f; } u; u.i = i; return u.f; }

// DPP16 lane permute within rows of 16 (pure VALU, no LDS)
template <int CTRL>
DEV float dppf(float v) {
  return i2f(__builtin_amdgcn_update_dpp(0, f2i(v), CTRL, 0xF, 0xF, false));
}
// Butterfly reduction across 16 lanes (half-wave groups stay independent):
// xor1 = quad_perm(1,0,3,2)=0xB1, xor2 = quad_perm(2,3,0,1)=0x4E,
// then row_half_mirror(0x141) ~ xor4, row_mirror(0x140) ~ xor8.
DEV float row16_max(float x) {
  x = fmaxf(x, dppf<0xB1>(x));
  x = fmaxf(x, dppf<0x4E>(x));
  x = fmaxf(x, dppf<0x141>(x));
  x = fmaxf(x, dppf<0x140>(x));
  return x;
}
DEV float row16_sum(float x) {
  x += dppf<0xB1>(x);
  x += dppf<0x4E>(x);
  x += dppf<0x141>(x);
  x += dppf<0x140>(x);
  return x;
}

DEV v16bf cat8(v8bf lo, v8bf hi) {
  v16bf r;
#pragma unroll
  for (int i = 0; i < 8; ++i) { r[i] = lo[i]; r[i + 8] = hi[i]; }
  return r;
}

DEV v8f wmma_bf16(v16bf a, v16bf b, v8f c) {
  // D = A(16x32 bf16) * B(32x16 bf16) + C(16x16 f32)
  return __builtin_amdgcn_wmma_f32_16x16x32_bf16(
      false, a, false, b, (short)0, c, false, false);
}

// Problem constants (match reference)
constexpr int Bc  = 2;
constexpr int Tc  = 2048;
constexpr int Dc  = 2048;   // D_IN == D_OUT
constexpr int Hc  = 32;
constexpr int KVc = 8;
constexpr int HDc = 64;

// ---------------------------------------------------------------------------
// Elementwise fp32 -> bf16
// ---------------------------------------------------------------------------
__global__ void k_cvt(const float* __restrict__ in, u16* __restrict__ out, int n) {
  int i = blockIdx.x * blockDim.x + threadIdx.x;
  if (i < n) out[i] = f2b(in[i]);
}

// W (K x N fp32, row-major) -> Wt (N x K bf16, row-major)
__global__ void k_cvt_wT(const float* __restrict__ W, u16* __restrict__ Wt,
                         int K, int N) {
  int i = blockIdx.x * blockDim.x + threadIdx.x;
  int total = K * N;
  if (i >= total) return;
  int kk = i / N, nn = i - kk * N;
  Wt[(size_t)nn * K + kk] = f2b(W[i]);
}

// ---------------------------------------------------------------------------
// GEMM: C[MxN] = A[MxK]bf16 @ Bt[NxK]bf16 (B pre-transposed).
// Block tile 64(M) x 128(N), BK=32, 256 threads = 8 waves (4x2 wave grid),
// each wave computes a 16x64 strip via 4 WMMA accumulators.
// ---------------------------------------------------------------------------
template <bool OUT_BF16>
__global__ __launch_bounds__(256) void k_gemm(const u16* __restrict__ A,
                                              const u16* __restrict__ Bt,
                                              void* __restrict__ Cout,
                                              int M, int N, int K) {
  __shared__ u16 sA[64 * 32];
  __shared__ u16 sB[128 * 32];

  const int tid  = threadIdx.x;
  const int wave = tid >> 5;
  const int lane = tid & 31;
  const int lg = lane >> 4, ln = lane & 15;
  const int wm = wave >> 1;          // 0..3 -> M strip
  const int wn = wave & 1;           // 0..1 -> N strip
  const int m0 = blockIdx.y * 64;
  const int n0 = blockIdx.x * 128;

  v8f acc[4];
#pragma unroll
  for (int nt = 0; nt < 4; ++nt)
#pragma unroll
    for (int r = 0; r < 8; ++r) acc[nt][r] = 0.f;

  const int am = tid >> 2, ach = tid & 3;   // A-tile loader coords

  for (int kk = 0; kk < K; kk += 32) {
    // Stage A tile: 64 x 32 bf16 (one 16B chunk per thread)
    *(uint4*)&sA[am * 32 + ach * 8] =
        *(const uint4*)&A[(size_t)(m0 + am) * K + kk + ach * 8];
    // Stage Bt tile: 128 x 32 bf16 (two 16B chunks per thread)
#pragma unroll
    for (int i = 0; i < 2; ++i) {
      int idx = tid + i * 256;
      int n = idx >> 2, ch = idx & 3;
      *(uint4*)&sB[n * 32 + ch * 8] =
          *(const uint4*)&Bt[(size_t)(n0 + n) * K + kk + ch * 8];
    }
    if (kk + 32 < K) {
      __builtin_prefetch(&A[(size_t)(m0 + am) * K + kk + 32 + ach * 8], 0, 0);
      __builtin_prefetch(&Bt[(size_t)(n0 + (tid >> 2)) * K + kk + 32], 0, 0);
    }
    __syncthreads();

    // A fragment (16x32): lane group g holds K = {16g..16g+15}
    const int row = wm * 16 + ln;
    const int bk = lg * 16;
    v16bf a = *(const v16bf*)&sA[row * 32 + bk];
#pragma unroll
    for (int nt = 0; nt < 4; ++nt) {
      // B fragment (32x16): lane = column, 16 contiguous K values
      const int n = wn * 64 + nt * 16 + ln;
      v16bf b = *(const v16bf*)&sB[n * 32 + lg * 16];
      acc[nt] = wmma_bf16(a, b, acc[nt]);
    }
    __syncthreads();
  }

  // Writeback per C/D layout: VGPR r, lane -> row r + 8*(lane/16), col lane%16
#pragma unroll
  for (int nt = 0; nt < 4; ++nt) {
#pragma unroll
    for (int r = 0; r < 8; ++r) {
      const int row = m0 + wm * 16 + r + lg * 8;
      const int col = n0 + wn * 64 + nt * 16 + ln;
      if (OUT_BF16)
        ((u16*)Cout)[(size_t)row * N + col] = f2b(acc[nt][r]);
      else
        ((float*)Cout)[(size_t)row * N + col] = acc[nt][r];
    }
  }
}

// ---------------------------------------------------------------------------
// RoPE + head transpose: in (B,T,heads,64) bf16 -> out (B,heads,T,64) bf16
// Llama-style adjacent-pair rotation; one thread per pair.
// ---------------------------------------------------------------------------
__global__ void k_rope(const u16* __restrict__ in, u16* __restrict__ out,
                       int heads, int npairs) {
  int idx = blockIdx.x * blockDim.x + threadIdx.x;
  if (idx >= npairs) return;
  int i = idx & 31; int p = idx >> 5;
  int h = p % heads; p /= heads;
  int t = p % Tc;    int b = p / Tc;
  float inv = __powf(10000.0f, -(float)(2 * i) * (1.0f / 64.0f));
  float ang = (float)t * inv;
  float s, c; __sincosf(ang, &s, &c);
  size_t src = ((((size_t)b * Tc + t) * heads + h) << 6) + 2 * i;
  float x0 = b2f(in[src]), x1 = b2f(in[src + 1]);
  size_t dst = ((((size_t)b * heads + h) * Tc + t) << 6) + 2 * i;
  out[dst]     = f2b(x0 * c - x1 * s);
  out[dst + 1] = f2b(x0 * s + x1 * c);
}

// v: (B,T,KV,64) bf16 -> vt: (B,KV,64,T) bf16 (so p@v B-frags are contiguous)
__global__ void k_vtrans(const u16* __restrict__ in, u16* __restrict__ out,
                         int n) {
  int idx = blockIdx.x * blockDim.x + threadIdx.x;
  if (idx >= n) return;
  int t = idx % Tc; int r = idx / Tc;
  int hd = r % HDc; r /= HDc;
  int kv = r % KVc; int b = r / KVc;
  out[idx] = in[(((size_t)b * Tc + t) * KVc + kv) * HDc + hd];
}

// ---------------------------------------------------------------------------
// Flash attention (causal, GQA). 4 waves/block; each wave owns a 16-row
// q tile; loops over 32-key blocks: two score WMMAs per 16-key subtile
// (K-dim = HD = 64 -> 2x 16x16x32), online softmax with DPP16 row
// reductions, exp(P) restaged via per-wave LDS into A-fragment layout,
// then 4 WMMAs for p@v (N = HD = 64).
// ---------------------------------------------------------------------------
__global__ __launch_bounds__(128) void k_attn(const u16* __restrict__ q,
                                              const u16* __restrict__ k,
                                              const u16* __restrict__ vt,
                                              u16* __restrict__ ctx) {
  __shared__ u16 sP[4][16 * 32];
  const int wave = threadIdx.x >> 5, lane = threadIdx.x & 31;
  const int lg = lane >> 4, ln = lane & 15;
  const int qb = blockIdx.x & 31;          // T/64 tiles
  const int h  = (blockIdx.x >> 5) & 31;   // H
  const int b  = blockIdx.x >> 10;
  const int kvh = h >> 2;                  // G = H/KV = 4
  const int qrow0 = qb * 64 + wave * 16;

  // Q fragments (held for whole kernel): A layout, K-dim = hd
  const u16* qrow = q + (((size_t)(b * Hc + h)) * Tc + (qrow0 + ln)) * HDc;
  const int bk = lg * 16;
  v16bf aq0 = *(const v16bf*)&qrow[bk];
  v16bf aq1 = *(const v16bf*)&qrow[bk + 32];

  v8f O[4];
  float rmax[8], rsum[8];
#pragma unroll
  for (int nt = 0; nt < 4; ++nt)
#pragma unroll
    for (int r = 0; r < 8; ++r) O[nt][r] = 0.f;
#pragma unroll
  for (int r = 0; r < 8; ++r) { rmax[r] = -1e30f; rsum[r] = 0.f; }

  const float scale = 0.125f;  // 1/sqrt(64)
  u16* sPw = sP[wave];
  const u16* kbase = k  + ((size_t)(b * KVc + kvh)) * Tc * HDc;
  const u16* vbase = vt + ((size_t)(b * KVc + kvh)) * HDc * Tc;

  const int nkb = (qrow0 + 47) >> 5;  // 32-key blocks covering rows <= qrow0+15
  for (int jb = 0; jb < nkb; ++jb) {
    const int kb0 = jb * 32;
    float S[2][8];
#pragma unroll
    for (int st = 0; st < 2; ++st) {
      const int key = kb0 + st * 16 + ln;
      const u16* krow = kbase + (size_t)key * HDc;
      const int hdb = lg * 16;
      v16bf bk0 = *(const v16bf*)&krow[hdb];        // hd 0..31 slice
      v16bf bk1 = *(const v16bf*)&krow[32 + hdb];   // hd 32..63 slice
      v8f s;
#pragma unroll
      for (int r = 0; r < 8; ++r) s[r] = 0.f;
      s = wmma_bf16(aq0, bk0, s);
      s = wmma_bf16(aq1, bk1, s);
#pragma unroll
      for (int r = 0; r < 8; ++r) {
        const int row = qrow0 + r + lg * 8;
        S[st][r] = (key > row) ? -1e30f : s[r] * scale;
      }
    }
    // Online softmax over this 32-key block; row stats live across 16 lanes,
    // reduced with DPP16 butterflies (pure VALU, co-executes with WMMA).
#pragma unroll
    for (int r = 0; r < 8; ++r) {
      const float m2 = row16_max(fmaxf(S[0][r], S[1][r]));
      const float nm = fmaxf(rmax[r], m2);
      const float f  = __expf(rmax[r] - nm);
      const float p0 = __expf(S[0][r] - nm);
      const float p1 = __expf(S[1][r] - nm);
      const float rs = row16_sum(p0 + p1);
      rsum[r] = rsum[r] * f + rs;
      rmax[r] = nm;
      const int prow = r + lg * 8;
      sPw[prow * 32 + ln]      = f2b(p0);
      sPw[prow * 32 + 16 + ln] = f2b(p1);
#pragma unroll
      for (int nt = 0; nt < 4; ++nt) O[nt][r] *= f;
    }
    __builtin_amdgcn_wave_barrier();  // order LDS stores vs. fragment reloads

    // P (16x32) as A-fragment, V block (32 keys x 64 hd) as 4 B-fragments
    v16bf ap = *(const v16bf*)&sPw[ln * 32 + bk];
#pragma unroll
    for (int nt = 0; nt < 4; ++nt) {
      const int hd = nt * 16 + ln;
      v16bf bv = *(const v16bf*)&vbase[(size_t)hd * Tc + kb0 + lg * 16];
      O[nt] = wmma_bf16(ap, bv, O[nt]);
    }
  }

  // Normalize and write ctx (B,T,H*HD) bf16
#pragma unroll
  for (int r = 0; r < 8; ++r) {
    const float inv = 1.0f / rsum[r];
    const int t = qrow0 + r + lg * 8;
    u16* crow = ctx + ((size_t)b * Tc + t) * (Hc * HDc) + h * HDc;
#pragma unroll
    for (int nt = 0; nt < 4; ++nt) crow[nt * 16 + ln] = f2b(O[nt][r] * inv);
  }
}

// ---------------------------------------------------------------------------
// Launch
// ---------------------------------------------------------------------------
extern "C" void kernel_launch(void* const* d_in, const int* in_sizes, int n_in,
                              void* d_out, int out_size, void* d_ws, size_t ws_size,
                              hipStream_t stream) {
  (void)in_sizes; (void)n_in; (void)out_size; (void)ws_size;
  const float* x  = (const float*)d_in[0];
  const float* Wq = (const float*)d_in[1];
  const float* Wk = (const float*)d_in[2];
  const float* Wv = (const float*)d_in[3];
  const float* Wo = (const float*)d_in[4];
  float* out = (float*)d_out;

  const int M = Bc * Tc;          // 4096 token rows
  char* ws = (char*)d_ws;
  size_t o = 0;
  auto alloc = [&](size_t bytes) { char* p = ws + o; o += bytes; return p; };
  u16* xb    = (u16*)alloc((size_t)M * Dc * 2);          // 16 MB
  u16* WqT   = (u16*)alloc((size_t)Dc * Dc * 2);         //  8 MB
  u16* WkT   = (u16*)alloc((size_t)(KVc*HDc) * Dc * 2);  //  2 MB
  u16* WvT   = (u16*)alloc((size_t)(KVc*HDc) * Dc * 2);  //  2 MB
  u16* WoT   = (u16*)alloc((size_t)Dc * Dc * 2);         //  8 MB
  u16* q_pre = (u16*)alloc((size_t)M * Dc * 2);          // 16 MB
  u16* k_pre = (u16*)alloc((size_t)M * KVc * HDc * 2);   //  4 MB
  u16* v_pre = (u16*)alloc((size_t)M * KVc * HDc * 2);   //  4 MB
  u16* q_rot = (u16*)alloc((size_t)M * Dc * 2);          // 16 MB
  u16* k_rot = (u16*)alloc((size_t)M * KVc * HDc * 2);   //  4 MB
  u16* v_t   = (u16*)alloc((size_t)M * KVc * HDc * 2);   //  4 MB
  u16* ctxb  = (u16*)alloc((size_t)M * Dc * 2);          // 16 MB (~100 MB total)

  const int nx = M * Dc;
  k_cvt<<<(nx + 255) / 256, 256, 0, stream>>>(x, xb, nx);
  k_cvt_wT<<<(Dc * Dc + 255) / 256, 256, 0, stream>>>(Wq, WqT, Dc, Dc);
  k_cvt_wT<<<(Dc * KVc * HDc + 255) / 256, 256, 0, stream>>>(Wk, WkT, Dc, KVc * HDc);
  k_cvt_wT<<<(Dc * KVc * HDc + 255) / 256, 256, 0, stream>>>(Wv, WvT, Dc, KVc * HDc);
  k_cvt_wT<<<(Dc * Dc + 255) / 256, 256, 0, stream>>>(Wo, WoT, Dc, Dc);

  // QKV projections (bf16 WMMA, f32 accumulate, bf16 out)
  k_gemm<true><<<dim3(Dc / 128, M / 64), 256, 0, stream>>>(xb, WqT, q_pre, M, Dc, Dc);
  k_gemm<true><<<dim3((KVc * HDc) / 128, M / 64), 256, 0, stream>>>(xb, WkT, k_pre, M, KVc * HDc, Dc);
  k_gemm<true><<<dim3((KVc * HDc) / 128, M / 64), 256, 0, stream>>>(xb, WvT, v_pre, M, KVc * HDc, Dc);

  // RoPE + layout transforms
  const int qpairs = M * Hc * (HDc / 2);
  const int kpairs = M * KVc * (HDc / 2);
  k_rope<<<(qpairs + 255) / 256, 256, 0, stream>>>(q_pre, q_rot, Hc, qpairs);
  k_rope<<<(kpairs + 255) / 256, 256, 0, stream>>>(k_pre, k_rot, KVc, kpairs);
  const int nv = M * KVc * HDc;
  k_vtrans<<<(nv + 255) / 256, 256, 0, stream>>>(v_pre, v_t, nv);

  // Causal flash attention
  k_attn<<<Bc * Hc * (Tc / 64), 128, 0, stream>>>(q_rot, k_rot, v_t, ctxb);

  // Output projection -> fp32 result
  k_gemm<false><<<dim3(Dc / 128, M / 64), 256, 0, stream>>>(ctxb, WoT, out, M, Dc, Dc);
}
